// FSAM_80925773791358
// MI455X (gfx1250) — matrix operand order repaired and checked by reading, not compiled, not run on hardware
//
#include <hip/hip_runtime.h>
#include <hip/hip_bf16.h>
#include <stdint.h>

#define NJ 50176          // D*H*W = 16*56*56
#define CHIN 256
#define CHAL 128
#define NMF_L 8
#define NV_ELEMS 6422528  // 128*50176
#define NH_ELEMS 401408   // 8*50176
#define EPSF 1.1920929e-07f

typedef __attribute__((ext_vector_type(16))) _Float16 v16h;
typedef __attribute__((ext_vector_type(8)))  float    v8f;

// K offset inside a 16x32 f16 A fragment for (lane>>4, element): contiguous k = 16*hi + e (matches B staging).
__device__ inline int koff_a(int hi, int e) {
  return (hi << 4) + e;
}

// ---- One-time weight conversion: fp32 row-major -> f16 A-fragment order ----
// dst index = ((mt*nks + ks)*32 + lane)*16 + e
__global__ __launch_bounds__(256) void k_convw(const float* __restrict__ src,
                                               _Float16* __restrict__ dst,
                                               int nks, int lda) {
  const int idx = blockIdx.x * 256 + threadIdx.x;
  const int e = idx & 15;
  const int lane = (idx >> 4) & 31;
  const int t = idx >> 9;
  const int ks = t % nks;
  const int mt = t / nks;
  const int K = ks * 32 + koff_a(lane >> 4, e);
  const int row = mt * 16 + (lane & 15);
  dst[idx] = (_Float16)src[row * lda + K];
}

// ---- K1: V = relu(w_pre @ eps + b)  (128x50176 = [128x256]x[256x50176]) ----
__global__ __launch_bounds__(256) void k_pre(const float* __restrict__ eps,
                                             const _Float16* __restrict__ Af,
                                             const float* __restrict__ b_pre,
                                             float* __restrict__ V) {
  __shared__ __align__(32) _Float16 sB[8 * 512];  // 8 K-steps, fragment order
  const int tid = threadIdx.x;
  const int colBase = blockIdx.x * 16;
  // stage eps tile (256 K x 16 cols), fp32->f16, directly into fragment layout
  for (int idx = tid; idx < 4096; idx += 256) {
    const int k = idx >> 4, c = idx & 15;
    const int ks = k >> 5, r = k & 31, hi = r >> 4, e = r & 15;
    sB[ks * 512 + ((hi << 4) | c) * 16 + e] = (_Float16)eps[k * NJ + colBase + c];
  }
  __syncthreads();
  const int wave = tid >> 5, lane = tid & 31;
  v8f c = {};
#pragma unroll
  for (int ks = 0; ks < 8; ++ks) {
    v16h a = *(const v16h*)&Af[((wave * 8 + ks) * 32 + lane) * 16];
    v16h b = *(const v16h*)&sB[(ks * 32 + lane) * 16];
    c = __builtin_amdgcn_wmma_f32_16x16x32_f16(false, a, false, b, (short)0, c, false, false);
  }
#pragma unroll
  for (int v = 0; v < 8; ++v) {
    const int row = wave * 16 + v + ((lane >> 4) << 3);
    const int col = colBase + (lane & 15);
    float val = c[v] + b_pre[row];
    V[row * NJ + col] = val > 0.f ? val : 0.f;
  }
}

// ---- mean(V) reduction ----
__global__ __launch_bounds__(256) void k_mean(const float* __restrict__ V, float* __restrict__ acc) {
  __shared__ float red[256];
  const int tid = threadIdx.x;
  float s = 0.f;
  for (int i = blockIdx.x * 256 + tid; i < NV_ELEMS; i += gridDim.x * 256) s += V[i];
  red[tid] = s;
  __syncthreads();
  for (int o = 128; o > 0; o >>= 1) { if (tid < o) red[tid] += red[tid + o]; __syncthreads(); }
  if (tid == 0) atomicAdd(acc, red[0]);
}

__device__ inline uint32_t hashu(uint32_t x) {
  x ^= x >> 16; x *= 0x7feb352du; x ^= x >> 15; x *= 0x846ca68bu; x ^= x >> 16; return x;
}

// ---- NMF init: W,H = |avg * noise|, avg = sqrt(mean(V)/L) ----
__global__ __launch_bounds__(256) void k_init(const float* __restrict__ meanAcc,
                                              float* __restrict__ W, float* __restrict__ H) {
  const int i = blockIdx.x * 256 + threadIdx.x;
  const float avg = sqrtf(fmaxf(meanAcc[0] / (float)NV_ELEMS, 0.f) / (float)NMF_L);
  if (i < NH_ELEMS) {
    const float u = (hashu(0x9E3779B9u ^ (uint32_t)i) >> 8) * (1.f / 16777216.f);
    H[i] = avg * (0.1f + 1.9f * u);
  }
  if (i < CHAL * NMF_L) {
    const float u = (hashu(0xB5297A4Du ^ (uint32_t)i) >> 8) * (1.f / 16777216.f);
    W[i] = avg * (0.1f + 1.9f * u);
  }
}

// ---- NMF iteration kernel 1: H *= (W^T V)/(W^T W H + eps); accumulate VH^T, HH^T ----
__global__ __launch_bounds__(256) void k_h_update(const float* __restrict__ V,
                                                  float* __restrict__ H,
                                                  const float* __restrict__ W,
                                                  float* __restrict__ VHt,
                                                  float* __restrict__ HHt) {
  __shared__ float sW[CHAL * NMF_L];
  __shared__ float sWtW[NMF_L * NMF_L];
  __shared__ float sH[256 * NMF_L];
  const int tid = threadIdx.x;
  for (int idx = tid; idx < CHAL * NMF_L; idx += 256) sW[idx] = W[idx];
  __syncthreads();
  if (tid < 64) {
    const int a = tid >> 3, b = tid & 7;
    float s = 0.f;
    for (int r = 0; r < CHAL; ++r) s += sW[r * 8 + a] * sW[r * 8 + b];
    sWtW[tid] = s;
  }
  __syncthreads();
  const int j = blockIdx.x * 256 + tid;
  float u[8] = {0.f, 0.f, 0.f, 0.f, 0.f, 0.f, 0.f, 0.f};
  for (int r = 0; r < CHAL; ++r) {
    if (r + 8 < CHAL) __builtin_prefetch(&V[(r + 8) * NJ + j], 0, 1);  // global_prefetch_b8
    const float v = V[r * NJ + j];
#pragma unroll
    for (int l = 0; l < 8; ++l) u[l] += v * sW[r * 8 + l];
  }
  float h[8];
#pragma unroll
  for (int l = 0; l < 8; ++l) h[l] = H[l * NJ + j];
#pragma unroll
  for (int l = 0; l < 8; ++l) {
    float s = 0.f;
#pragma unroll
    for (int m = 0; m < 8; ++m) s += sWtW[l * 8 + m] * h[m];
    const float hn = h[l] * u[l] / (s + EPSF);
    H[l * NJ + j] = hn;
    sH[tid * 8 + l] = hn;
  }
  __syncthreads();
  // VH^T partials: thread owns (r = tid>>1, 4 l's); V chunk is L1-hot from scan above.
  const int r = tid >> 1, l0 = (tid & 1) * 4;
  const int jb = blockIdx.x * 256;
  float a0 = 0.f, a1 = 0.f, a2 = 0.f, a3 = 0.f;
  for (int jj = 0; jj < 256; ++jj) {
    const float v = V[r * NJ + jb + jj];
    a0 += v * sH[jj * 8 + l0];
    a1 += v * sH[jj * 8 + l0 + 1];
    a2 += v * sH[jj * 8 + l0 + 2];
    a3 += v * sH[jj * 8 + l0 + 3];
  }
  atomicAdd(&VHt[r * 8 + l0], a0);
  atomicAdd(&VHt[r * 8 + l0 + 1], a1);
  atomicAdd(&VHt[r * 8 + l0 + 2], a2);
  atomicAdd(&VHt[r * 8 + l0 + 3], a3);
  if (tid < 64) {
    const int a = tid >> 3, b = tid & 7;
    float s = 0.f;
    for (int jj = 0; jj < 256; ++jj) s += sH[jj * 8 + a] * sH[jj * 8 + b];
    atomicAdd(&HHt[tid], s);
  }
}

// ---- NMF iteration kernel 2: W *= VH^T/(W HH^T + eps); re-zero accumulators ----
__global__ __launch_bounds__(1024) void k_w_update(float* __restrict__ W,
                                                   float* __restrict__ VHt,
                                                   float* __restrict__ HHt) {
  __shared__ float sW[CHAL * NMF_L];
  __shared__ float sHHt[64];
  const int tid = threadIdx.x;  // 1024 threads == 128x8 entries
  sW[tid] = W[tid];
  if (tid < 64) sHHt[tid] = HHt[tid];
  __syncthreads();
  const int r = tid >> 3, l = tid & 7;
  float s = 0.f;
#pragma unroll
  for (int m = 0; m < 8; ++m) s += sW[r * 8 + m] * sHHt[m * 8 + l];
  W[tid] = sW[tid] * VHt[tid] / (s + EPSF);
  VHt[tid] = 0.f;
  if (tid < 64) HHt[tid] = 0.f;
}

// ---- Vhat = W@H (stored f16 for post GEMMs) + ||V - Vhat||^2 accumulation ----
__global__ __launch_bounds__(256) void k_vhat_err(const float* __restrict__ V,
                                                  const float* __restrict__ H,
                                                  const float* __restrict__ W,
                                                  _Float16* __restrict__ Vh,
                                                  float* __restrict__ errAcc) {
  __shared__ float red[256];
  const int tid = threadIdx.x;
  const int r = blockIdx.y;
  const int j = blockIdx.x * 256 + tid;
  float vhat = 0.f;
#pragma unroll
  for (int l = 0; l < 8; ++l) vhat += W[r * 8 + l] * H[l * NJ + j];
  Vh[r * NJ + j] = (_Float16)vhat;
  const float d = V[r * NJ + j] - vhat;
  red[tid] = d * d;
  __syncthreads();
  for (int o = 128; o > 0; o >>= 1) { if (tid < o) red[tid] += red[tid + o]; __syncthreads(); }
  if (tid == 0) atomicAdd(errAcc, red[0]);
}

// ---- Fused post chain: out = w_post2 @ relu(w_post1 @ Vhat), WMMA + async LDS copy ----
__global__ __launch_bounds__(256) void k_post(const _Float16* __restrict__ Vh,
                                              const _Float16* __restrict__ A1,
                                              const _Float16* __restrict__ A2,
                                              float* __restrict__ out) {
  __shared__ __align__(32) _Float16 sRaw[CHAL * 16];  // Vhat tile, row-major (4KB)
  __shared__ __align__(32) _Float16 sBv[4 * 512];     // Vhat tile, B-fragment order
  __shared__ __align__(32) _Float16 sYf[4 * 512];     // y tile, B-fragment order
  const int tid = threadIdx.x, wave = tid >> 5, lane = tid & 31;
  const int colBase = blockIdx.x * 16;

  // Async copy of the 4KB f16 Vhat tile into LDS: one b128 per thread.
  // GV mode: VDST = LDS byte offset (low 32 bits of generic LDS pointer), VADDR = 64-bit addr.
  {
    const int r = tid >> 1, part = tid & 1;
    const _Float16* gp = &Vh[r * NJ + colBase + part * 8];
    const unsigned long long ga = (unsigned long long)(uintptr_t)gp;
    const unsigned lo = (unsigned)(uintptr_t)(&sRaw[r * 16 + part * 8]);
    asm volatile("global_load_async_to_lds_b128 %0, %1, off"
                 :: "v"(lo), "v"(ga) : "memory");
  }
  asm volatile("s_wait_asynccnt 0x0" ::: "memory");
  __syncthreads();
  // Reorder row-major tile -> fragment order (LDS->LDS through registers).
  for (int idx = tid; idx < 2048; idx += 256) {
    const int k = idx >> 4, c = idx & 15;
    const int ks = k >> 5, r = k & 31, hi = r >> 4, e = r & 15;
    sBv[ks * 512 + ((hi << 4) | c) * 16 + e] = sRaw[k * 16 + c];
  }
  __syncthreads();

  // stage 1: y = relu(w_post1 @ Vhat_tile); write y straight into fragment order
  {
    v8f c = {};
#pragma unroll
    for (int ks = 0; ks < 4; ++ks) {
      v16h a = *(const v16h*)&A1[((wave * 4 + ks) * 32 + lane) * 16];
      v16h b = *(const v16h*)&sBv[(ks * 32 + lane) * 16];
      c = __builtin_amdgcn_wmma_f32_16x16x32_f16(false, a, false, b, (short)0, c, false, false);
    }
#pragma unroll
    for (int v = 0; v < 8; ++v) {
      const int row = wave * 16 + v + ((lane >> 4) << 3);
      const int col = lane & 15;
      const float val = c[v] > 0.f ? c[v] : 0.f;
      const int ks2 = row >> 5, r2 = row & 31, hi2 = r2 >> 4, e2 = r2 & 15;
      sYf[ks2 * 512 + ((hi2 << 4) | col) * 16 + e2] = (_Float16)val;
    }
  }
  __syncthreads();

  // stage 2: out = w_post2 @ y; hoist the 4 y B-fragments, reuse for both M passes
  v16h by[4];
#pragma unroll
  for (int ks = 0; ks < 4; ++ks) by[ks] = *(const v16h*)&sYf[(ks * 32 + lane) * 16];
#pragma unroll
  for (int rep = 0; rep < 2; ++rep) {
    const int mt = wave + rep * 8;
    v8f c = {};
#pragma unroll
    for (int ks = 0; ks < 4; ++ks) {
      v16h a = *(const v16h*)&A2[((mt * 4 + ks) * 32 + lane) * 16];
      c = __builtin_amdgcn_wmma_f32_16x16x32_f16(false, a, false, by[ks], (short)0, c, false, false);
    }
#pragma unroll
    for (int v = 0; v < 8; ++v) {
      const int row = mt * 16 + v + ((lane >> 4) << 3);
      out[row * NJ + colBase + (lane & 15)] = c[v];
    }
  }
}

__global__ void k_final(const float* __restrict__ errAcc, float* __restrict__ out) {
  if (threadIdx.x == 0) out[CHIN * NJ] = sqrtf(errAcc[0]);
}

extern "C" void kernel_launch(void* const* d_in, const int* in_sizes, int n_in,
                              void* d_out, int out_size, void* d_ws, size_t ws_size,
                              hipStream_t stream) {
  const float* eps     = (const float*)d_in[0];
  const float* w_pre   = (const float*)d_in[1];
  const float* b_pre   = (const float*)d_in[2];
  const float* w_post1 = (const float*)d_in[3];
  const float* w_post2 = (const float*)d_in[4];
  float* out = (float*)d_out;

  // ws layout (float offsets):
  // 0: acc[2] | 2: VHt[1024] | 1026: HHt[64] | 2048: W[1024]
  // 4096: Apre f16[32768] | 20480: A1 f16[16384] | 28672: A2 f16[32768]
  // 45056: H[401408] | 446464: V[6422528] | 6868992: Vh f16[6422528]
  char* ws = (char*)d_ws;
  float* acc = (float*)ws;
  float* VHt = acc + 2;
  float* HHt = VHt + 1024;
  float* W    = (float*)(ws + (size_t)2048 * 4);
  _Float16* Apre = (_Float16*)(ws + (size_t)4096 * 4);
  _Float16* A1f  = (_Float16*)(ws + (size_t)20480 * 4);
  _Float16* A2f  = (_Float16*)(ws + (size_t)28672 * 4);
  float* H    = (float*)(ws + (size_t)45056 * 4);
  float* V    = (float*)(ws + (size_t)446464 * 4);
  _Float16* Vh = (_Float16*)(ws + (size_t)6868992 * 4);

  hipMemsetAsync(acc, 0, (2 + 1024 + 64) * sizeof(float), stream);

  // one-time weight -> f16 fragment-order conversion (L2-resident afterwards)
  k_convw<<<128, 256, 0, stream>>>(w_pre,   Apre, 8, CHIN);   // 8 Mtiles x 8 Ksteps
  k_convw<<<64,  256, 0, stream>>>(w_post1, A1f,  4, CHAL);   // 8 Mtiles x 4 Ksteps
  k_convw<<<128, 256, 0, stream>>>(w_post2, A2f,  4, CHAL);   // 16 Mtiles x 4 Ksteps

  k_pre<<<NJ / 16, 256, 0, stream>>>(eps, Apre, b_pre, V);
  k_mean<<<1024, 256, 0, stream>>>(V, &acc[0]);
  k_init<<<(NH_ELEMS + 255) / 256, 256, 0, stream>>>(&acc[0], W, H);

  for (int it = 0; it < 100; ++it) {
    k_h_update<<<NJ / 256, 256, 0, stream>>>(V, H, W, VHt, HHt);
    k_w_update<<<1, 1024, 0, stream>>>(W, VHt, HHt);
  }

  k_vhat_err<<<dim3(NJ / 256, CHAL), 256, 0, stream>>>(V, H, W, Vh, &acc[1]);
  k_post<<<NJ / 16, 256, 0, stream>>>(Vh, A1f, A2f, out);
  k_final<<<1, 32, 0, stream>>>(&acc[1], out);
}
